// GraphUnet_61186104099648
// MI455X (gfx1250) — compile-verified, matrix operand order checked by reading, not run on hardware
//
#include <hip/hip_runtime.h>

typedef __attribute__((ext_vector_type(16))) _Float16 v16h;
typedef __attribute__((ext_vector_type(8)))  _Float16 v8h;
typedef __attribute__((ext_vector_type(8)))  float    v8f;

#define TPB 256

union PkU { _Float16 h[2]; unsigned u; };
static __device__ __forceinline__ unsigned pk2(float a, float b) {
  PkU p; p.h[0] = (_Float16)a; p.h[1] = (_Float16)b; return p.u;
}

// ======================= WMMA GEMM (LDS-staged, K-step 64) =======================
// C[M,N] = act(A[M,K] @ B[K,N] + bias) + addend
// Block: 256 threads = 8 waves, 32(M) x 64(N) macro-tile, wave grid 2x4.
// K stepped by 64 -> two back-to-back v_wmma per barrier pair (XDL-friendly).
// A staged row-major f16 (stride 64h = 128B); B staged transposed [n][k] f16
// (stride 72h = 144B: 16B-aligned for ds_load_b128, bank-depadded).
// Each lane's fragment = two contiguous 16B runs -> 2x ds_load_b128 / operand.
// Interior tiles: scalar-uniform unguarded fast path; edges: clamped addresses
// + VALU selects (no EXEC-masked loads on the fast path).
template <typename TA>
__global__ __launch_bounds__(256) void gemm_wmma_kernel(
    const TA* __restrict__ A, int lda,
    const float* __restrict__ B, int ldb,
    const float* __restrict__ bias,
    const float* __restrict__ addend,
    float* __restrict__ C,
    int M, int N, int K, int relu) {
  __shared__ _Float16 lds_a[32 * 64];   // [m_local][k], row stride 64 halves
  __shared__ _Float16 lds_b[64 * 72];   // [n_local][k], row stride 72 halves
  unsigned* lds_a32 = (unsigned*)lds_a;
  unsigned* lds_b32 = (unsigned*)lds_b;

  const int lane = threadIdx.x;                 // 0..31 (wave32)
  const int w    = threadIdx.y;                 // 0..7
  const int t    = w * 32 + lane;               // 0..255
  const int m_blk = blockIdx.y * 32;
  const int n_blk = blockIdx.x * 64;
  const int wm = (w >> 2) * 16;                 // 0 or 16
  const int wn = (w & 3) * 16;                  // 0,16,32,48
  const int hi  = lane >> 4;                    // half-wave select
  const int l16 = lane & 15;

  v8f acc = {};
  for (int kb = 0; kb < K; kb += 64) {
    const bool a_full = (m_blk + 32 <= M) && (kb + 64 <= K);   // block-uniform
    const bool b_full = (n_blk + 64 <= N) && (kb + 64 <= K);   // block-uniform

    // ---- stage A: 32 rows x 64 k as 1024 packed dwords (4 per thread) ----
    if (a_full) {
#pragma unroll
      for (int i = 0; i < 4; ++i) {
        const int p = t + i * 256;                 // dword index == r*32+kp
        const int r = p >> 5, k2 = (p & 31) * 2;
        const size_t base = (size_t)(m_blk + r) * lda + kb + k2;
        lds_a32[p] = pk2((float)A[base], (float)A[base + 1]);
      }
    } else {
#pragma unroll
      for (int i = 0; i < 4; ++i) {
        const int p = t + i * 256;
        const int r = p >> 5, k2 = (p & 31) * 2;
        const int gm = m_blk + r, gk = kb + k2;
        const int gms = gm < M ? gm : M - 1;       // clamped, always in-bounds
        const int gk0 = gk < K ? gk : K - 1;
        const int gk1 = gk + 1 < K ? gk + 1 : K - 1;
        float v0 = (float)A[(size_t)gms * lda + gk0];
        float v1 = (float)A[(size_t)gms * lda + gk1];
        if (gm >= M || gk     >= K) v0 = 0.f;      // VALU select
        if (gm >= M || gk + 1 >= K) v1 = 0.f;
        lds_a32[p] = pk2(v0, v1);
      }
    }

    // ---- stage B transposed: 64 k x 64 n -> lds_b[n][k], 2048 pairs ----
    if (b_full) {
#pragma unroll
      for (int i = 0; i < 8; ++i) {
        const int p = t + i * 256;
        const int n = p & 63, kp = p >> 6;         // kp = 0..31
        const size_t base = (size_t)(kb + kp * 2) * ldb + n_blk + n;
        lds_b32[n * 36 + kp] = pk2(B[base], B[base + ldb]);
      }
    } else {
#pragma unroll
      for (int i = 0; i < 8; ++i) {
        const int p = t + i * 256;
        const int n = p & 63, kp = p >> 6;
        const int gn = n_blk + n, gk = kb + kp * 2;
        const int gns = gn < N ? gn : N - 1;
        const int gk0 = gk < K ? gk : K - 1;
        const int gk1 = gk + 1 < K ? gk + 1 : K - 1;
        float v0 = B[(size_t)gk0 * ldb + gns];
        float v1 = B[(size_t)gk1 * ldb + gns];
        if (gn >= N || gk     >= K) v0 = 0.f;
        if (gn >= N || gk + 1 >= K) v1 = 0.f;
        lds_b32[n * 36 + kp] = pk2(v0, v1);
      }
    }

    // prefetch next K-slab (clamped -> unconditional, speculative-safe)
    {
      const int nk  = (kb + 64 < K) ? kb + 64 : kb;
      const int gmp = m_blk + (t >> 3) < M ? m_blk + (t >> 3) : M - 1;
      __builtin_prefetch(&A[(size_t)gmp * lda + nk], 0, 1);
      const int gnp = n_blk + (t & 63) < N ? n_blk + (t & 63) : N - 1;
      const int gkp = nk + (t >> 6) * 16 < K ? nk + (t >> 6) * 16 : K - 1;
      __builtin_prefetch(&B[(size_t)gkp * ldb + gnp], 0, 1);
    }
    __syncthreads();

    // ---- branch-free fragments: 2x ds_load_b128/operand, 2 WMMAs back-to-back
    const v8h* pa = (const v8h*)&lds_a[(wm + l16) * 64];
    const v8h* pb = (const v8h*)&lds_b[(wn + l16) * 72];
#pragma unroll
    for (int kk = 0; kk < 2; ++kk) {
      const int o = kk * 4;                       // 32 halves = 4x v8h
      const v8h a0 = pa[o + hi], a1 = pa[o + 2 + hi];
      const v8h b0 = pb[o + hi], b1 = pb[o + 2 + hi];
      v16h af, bf;
#pragma unroll
      for (int i = 0; i < 8; ++i) {
        af[i] = a0[i]; af[i + 8] = a1[i];
        bf[i] = b0[i]; bf[i + 8] = b1[i];
      }
      acc = __builtin_amdgcn_wmma_f32_16x16x32_f16(false, af, false, bf,
                                                   (short)0, acc, false, false);
    }
    __syncthreads();
  }

#pragma unroll
  for (int r = 0; r < 8; ++r) {
    const int m = m_blk + wm + r + 8 * hi;        // ISA C/D layout
    const int n = n_blk + wn + l16;
    if (m < M && n < N) {
      float v = acc[r];
      if (bias)   v += bias[n];
      if (relu)   v = fmaxf(v, 0.f);
      if (addend) v += addend[(size_t)m * N + n];
      C[(size_t)m * N + n] = v;
    }
  }
}

// ======================= GAT helpers =======================

__global__ void attn_scores_kernel(const float* __restrict__ agg,
                                   const float* __restrict__ phi,
                                   int n, int d,
                                   float* __restrict__ si, float* __restrict__ sj) {
  const int i = blockIdx.x * blockDim.x + threadIdx.x;
  if (i >= n) return;
  const float* row = agg + (size_t)i * d;
  float a = 0.f, b = 0.f;
  for (int c = 0; c < d; ++c) {
    const float x = row[c];
    a += x * phi[c];
    b += x * phi[d + c];
  }
  si[i] = a;
  sj[i] = b;
}

// Row-wise masked softmax of H[i,j] = si[i]+sj[j] (mask: A[i,j]!=0 or j==i),
// emitted directly as f16 for the P @ agg WMMA GEMM.
__global__ void softmax_rows_kernel(const float* __restrict__ A,
                                    const float* __restrict__ si,
                                    const float* __restrict__ sj,
                                    _Float16* __restrict__ P, int n) {
  __shared__ float red[TPB];
  const int i = blockIdx.x, t = threadIdx.x;
  const float s_i = si[i];
  const float* Arow = A + (size_t)i * n;

  float lmax = -__builtin_inff();
  for (int j = t; j < n; j += TPB)
    if (Arow[j] != 0.f || j == i) lmax = fmaxf(lmax, s_i + sj[j]);
  red[t] = lmax; __syncthreads();
  for (int s = TPB / 2; s > 0; s >>= 1) {
    if (t < s) red[t] = fmaxf(red[t], red[t + s]);
    __syncthreads();
  }
  const float rmax = red[0]; __syncthreads();

  float lsum = 0.f;
  for (int j = t; j < n; j += TPB)
    if (Arow[j] != 0.f || j == i) lsum += __expf(s_i + sj[j] - rmax);
  red[t] = lsum; __syncthreads();
  for (int s = TPB / 2; s > 0; s >>= 1) {
    if (t < s) red[t] += red[t + s];
    __syncthreads();
  }
  const float inv = 1.f / red[0];

  for (int j = t; j < n; j += TPB) {
    const bool ok = (Arow[j] != 0.f || j == i);
    const float p = ok ? __expf(s_i + sj[j] - rmax) * inv : 0.f;
    P[(size_t)i * n + j] = (_Float16)p;
  }
}

// ======================= pool / unpool =======================

__global__ void pool_scores_kernel(const float* __restrict__ X,
                                   const float* __restrict__ w,
                                   const float* __restrict__ b,
                                   int n, int d, float* __restrict__ scores) {
  const int i = blockIdx.x * blockDim.x + threadIdx.x;
  if (i >= n) return;
  const float* row = X + (size_t)i * d;
  float s = 0.f;
  for (int c = 0; c < d; ++c) s += row[c] * w[c];
  s = (s + b[0]) * 0.01f;
  scores[i] = 1.f / (1.f + __expf(-s));
}

// Deterministic top-k by rank counting (desc by score, ties by index asc —
// matches lax.top_k ordering).
__global__ void topk_select_kernel(const float* __restrict__ scores, int n, int k,
                                   int* __restrict__ idx, float* __restrict__ vals) {
  const int i = blockIdx.x * blockDim.x + threadIdx.x;
  if (i >= n) return;
  const float s = scores[i];
  int rank = 0;
  for (int j = 0; j < n; ++j) {
    const float t = scores[j];
    rank += (t > s) || (t == s && j < i);
  }
  if (rank < k) { idx[rank] = i; vals[rank] = s; }
}

__global__ void gather_adj_kernel(const float* __restrict__ A, int n,
                                  const int* __restrict__ idx, int k,
                                  float* __restrict__ As) {
  const long t = (long)blockIdx.x * blockDim.x + threadIdx.x;
  if (t >= (long)k * k) return;
  const int a = (int)(t / k), b = (int)(t % k);
  As[t] = A[(size_t)idx[a] * n + idx[b]];
}

__global__ void gather_rows_kernel(const float* __restrict__ X,
                                   const int* __restrict__ idx,
                                   const float* __restrict__ vals,
                                   int k, int d, float* __restrict__ Xs) {
  const long t = (long)blockIdx.x * blockDim.x + threadIdx.x;
  if (t >= (long)k * d) return;
  const int a = (int)(t / d), c = (int)(t % d);
  Xs[t] = X[(size_t)idx[a] * d + c] * vals[a];
}

__global__ void zero_kernel(float* __restrict__ p, long cnt) {
  const long t = (long)blockIdx.x * blockDim.x + threadIdx.x;
  if (t < cnt) p[t] = 0.f;
}

__global__ void scatter_rows_kernel(const float* __restrict__ Xs,
                                    const int* __restrict__ idx,
                                    int k, int d, float* __restrict__ Xf) {
  const long t = (long)blockIdx.x * blockDim.x + threadIdx.x;
  if (t >= (long)k * d) return;
  const int a = (int)(t / d), c = (int)(t % d);
  Xf[(size_t)idx[a] * d + c] = Xs[t];
}

__global__ void concat_kernel(const float* __restrict__ Xa,
                              const float* __restrict__ Xb,
                              int n, int d, float* __restrict__ out) {
  const long t = (long)blockIdx.x * blockDim.x + threadIdx.x;
  if (t >= (long)n * 2 * d) return;
  const int i = (int)(t / (2 * d)), c = (int)(t % (2 * d));
  out[t] = (c < d) ? Xa[(size_t)i * d + c] : Xb[(size_t)i * d + (c - d)];
}

// ======================= host-side orchestration =======================

static inline int ceildiv(int a, int b) { return (a + b - 1) / b; }

static void gemm_f32(const float* A, int lda, const float* B, int ldb,
                     const float* bias, const float* addend, float* C,
                     int M, int N, int K, int relu, hipStream_t s) {
  dim3 blk(32, 8);
  dim3 grd(ceildiv(N, 64), ceildiv(M, 32));
  gemm_wmma_kernel<float><<<grd, blk, 0, s>>>(A, lda, B, ldb, bias, addend, C, M, N, K, relu);
}

static void gemm_f16A(const _Float16* A, int lda, const float* B, int ldb,
                      const float* bias, const float* addend, float* C,
                      int M, int N, int K, int relu, hipStream_t s) {
  dim3 blk(32, 8);
  dim3 grd(ceildiv(N, 64), ceildiv(M, 32));
  gemm_wmma_kernel<_Float16><<<grd, blk, 0, s>>>(A, lda, B, ldb, bias, addend, C, M, N, K, relu);
}

// out = relu(softmax_masked(si+sj^T) @ ((A@Xin)@W + b)) + addend
static void run_gat(const float* A, int n, const float* Xin, int fin,
                    const float* W, const float* b, const float* phi, int fout,
                    const float* addend, float* out,
                    float* t1, float* agg, float* si, float* sj, _Float16* P,
                    hipStream_t s) {
  gemm_f32(A, n, Xin, fin, nullptr, nullptr, t1, n, fin, n, 0, s);     // t1 = A @ X
  gemm_f32(t1, fin, W, fout, b, nullptr, agg, n, fout, fin, 0, s);     // agg = t1 @ W + b
  attn_scores_kernel<<<ceildiv(n, TPB), TPB, 0, s>>>(agg, phi, n, fout, si, sj);
  softmax_rows_kernel<<<n, TPB, 0, s>>>(A, si, sj, P, n);
  gemm_f16A(P, n, agg, fout, nullptr, addend, out, n, fout, n, 1, s);  // relu(P@agg)+addend
}

extern "C" void kernel_launch(void* const* d_in, const int* in_sizes, int n_in,
                              void* d_out, int out_size, void* d_ws, size_t ws_size,
                              hipStream_t stream) {
  (void)in_sizes; (void)n_in; (void)out_size; (void)ws_size;

  constexpr int N0 = 2048, D = 320;
  constexpr int K0 = 1843;  // int(0.9 * 2048)
  constexpr int K1 = 1290;  // int(0.7 * 1843)

  // JAX pytree (sorted-dict) flatten order of setup_inputs():
  const float* A0       = (const float*)d_in[0];
  const float* Xin      = (const float*)d_in[1];
  const float* bot_W    = (const float*)d_in[2];
  const float* bot_b    = (const float*)d_in[3];
  const float* bot_phi  = (const float*)d_in[4];
  const float* dn0_W    = (const float*)d_in[5];
  const float* dn0_b    = (const float*)d_in[6];
  const float* dn0_phi  = (const float*)d_in[7];
  const float* dn1_W    = (const float*)d_in[8];
  const float* dn1_b    = (const float*)d_in[9];
  const float* dn1_phi  = (const float*)d_in[10];
  const float* end_W    = (const float*)d_in[11];
  const float* end_b    = (const float*)d_in[12];
  const float* end_phi  = (const float*)d_in[13];
  const float* pool0_b  = (const float*)d_in[14];
  const float* pool0_w  = (const float*)d_in[15];
  const float* pool1_b  = (const float*)d_in[16];
  const float* pool1_w  = (const float*)d_in[17];
  const float* start_W  = (const float*)d_in[18];
  const float* start_b  = (const float*)d_in[19];
  const float* start_phi= (const float*)d_in[20];
  const float* up0_W    = (const float*)d_in[21];
  const float* up0_b    = (const float*)d_in[22];
  const float* up0_phi  = (const float*)d_in[23];
  const float* up1_W    = (const float*)d_in[24];
  const float* up1_b    = (const float*)d_in[25];
  const float* up1_phi  = (const float*)d_in[26];

  // d_out: [final X (2048x320) | start_outs (2048x320)]
  float* out_final = (float*)d_out;
  float* org       = (float*)d_out + (size_t)N0 * D;   // start_outs == org_X

  // workspace bump allocator
  char* wp = (char*)d_ws;
  auto alloc = [&](size_t bytes) -> void* {
    void* r = (void*)wp;
    wp += (bytes + 255) & ~(size_t)255;
    return r;
  };
  float*    t1     = (float*)   alloc((size_t)N0 * 2 * D * 4);   // up to 2048x640
  float*    agg    = (float*)   alloc((size_t)N0 * D * 4);
  float*    si     = (float*)   alloc((size_t)N0 * 4);
  float*    sj     = (float*)   alloc((size_t)N0 * 4);
  float*    scores = (float*)   alloc((size_t)N0 * 4);
  float*    vals   = (float*)   alloc((size_t)N0 * 4);
  int*      idx0   = (int*)     alloc((size_t)N0 * 4);
  int*      idx1   = (int*)     alloc((size_t)N0 * 4);
  _Float16* P      = (_Float16*)alloc((size_t)N0 * N0 * 2);
  float*    Xd0    = (float*)   alloc((size_t)N0 * D * 4);
  float*    Xd1    = (float*)   alloc((size_t)K0 * D * 4);
  float*    A1     = (float*)   alloc((size_t)K0 * K0 * 4);
  float*    A2     = (float*)   alloc((size_t)K1 * K1 * 4);
  float*    X1     = (float*)   alloc((size_t)K0 * D * 4);
  float*    X2     = (float*)   alloc((size_t)K1 * D * 4);
  float*    Xb     = (float*)   alloc((size_t)K1 * D * 4);
  float*    Xf1    = (float*)   alloc((size_t)K0 * D * 4);
  float*    Xu1    = (float*)   alloc((size_t)K0 * D * 4);
  float*    Xf2    = (float*)   alloc((size_t)N0 * D * 4);
  float*    Xu0    = (float*)   alloc((size_t)N0 * D * 4);
  float*    Xcat   = (float*)   alloc((size_t)N0 * 2 * D * 4);

  // ---- start GAT (also start_outs / org_X, written straight into d_out) ----
  run_gat(A0, N0, Xin, D, start_W, start_b, start_phi, D, nullptr, org,
          t1, agg, si, sj, P, stream);

  // ---- down 0 + pool 0 ----
  run_gat(A0, N0, org, D, dn0_W, dn0_b, dn0_phi, D, nullptr, Xd0,
          t1, agg, si, sj, P, stream);
  pool_scores_kernel<<<ceildiv(N0, TPB), TPB, 0, stream>>>(Xd0, pool0_w, pool0_b, N0, D, scores);
  topk_select_kernel<<<ceildiv(N0, TPB), TPB, 0, stream>>>(scores, N0, K0, idx0, vals);
  gather_adj_kernel<<<(int)(((long)K0 * K0 + TPB - 1) / TPB), TPB, 0, stream>>>(A0, N0, idx0, K0, A1);
  gather_rows_kernel<<<ceildiv(K0 * D, TPB), TPB, 0, stream>>>(Xd0, idx0, vals, K0, D, X1);

  // ---- down 1 + pool 1 ----
  run_gat(A1, K0, X1, D, dn1_W, dn1_b, dn1_phi, D, nullptr, Xd1,
          t1, agg, si, sj, P, stream);
  pool_scores_kernel<<<ceildiv(K0, TPB), TPB, 0, stream>>>(Xd1, pool1_w, pool1_b, K0, D, scores);
  topk_select_kernel<<<ceildiv(K0, TPB), TPB, 0, stream>>>(scores, K0, K1, idx1, vals);
  gather_adj_kernel<<<(int)(((long)K1 * K1 + TPB - 1) / TPB), TPB, 0, stream>>>(A1, K0, idx1, K1, A2);
  gather_rows_kernel<<<ceildiv(K1 * D, TPB), TPB, 0, stream>>>(Xd1, idx1, vals, K1, D, X2);

  // ---- bottom ----
  run_gat(A2, K1, X2, D, bot_W, bot_b, bot_phi, D, nullptr, Xb,
          t1, agg, si, sj, P, stream);

  // ---- up 0 (unpool to K0 graph) ----
  zero_kernel<<<ceildiv(K0 * D, TPB), TPB, 0, stream>>>(Xf1, (long)K0 * D);
  scatter_rows_kernel<<<ceildiv(K1 * D, TPB), TPB, 0, stream>>>(Xb, idx1, K1, D, Xf1);
  run_gat(A1, K0, Xf1, D, up0_W, up0_b, up0_phi, D, Xd1, Xu1,
          t1, agg, si, sj, P, stream);

  // ---- up 1 (unpool to full graph) ----
  zero_kernel<<<ceildiv(N0 * D, TPB), TPB, 0, stream>>>(Xf2, (long)N0 * D);
  scatter_rows_kernel<<<ceildiv(K0 * D, TPB), TPB, 0, stream>>>(Xu1, idx0, K0, D, Xf2);
  run_gat(A0, N0, Xf2, D, up1_W, up1_b, up1_phi, D, Xd0, Xu0,
          t1, agg, si, sj, P, stream);

  // ---- concat + end GAT (fin = 640) ----
  concat_kernel<<<ceildiv(N0 * 2 * D, TPB), TPB, 0, stream>>>(Xu0, org, N0, D, Xcat);
  run_gat(A0, N0, Xcat, 2 * D, end_W, end_b, end_phi, D, nullptr, out_final,
          t1, agg, si, sj, P, stream);
}
